// MANN_90434831385058
// MI455X (gfx1250) — matrix operand order repaired and checked by reading, not compiled, not run on hardware
//
#include <hip/hip_runtime.h>
#include <math.h>

typedef __bf16 bf16_t;
typedef __attribute__((ext_vector_type(16))) __bf16 v16bf;
typedef __attribute__((ext_vector_type(8)))  float  v8f;
typedef __attribute__((ext_vector_type(4)))  int    v4i;
typedef __attribute__((address_space(1))) v4i* gv4i_p;   // global int4*
typedef __attribute__((address_space(3))) v4i* lv4i_p;   // LDS int4*

#if defined(__HIP_DEVICE_COMPILE__) && __has_builtin(__builtin_amdgcn_global_load_async_to_lds_b128)
#define USE_ASYNC_LDS 1
#else
#define USE_ASYNC_LDS 0
#endif

namespace {
constexpr int B  = 1024, T = 50, IN = 784, N = 128, D = 40, H = 200, R = 4, C = 5;
constexpr int NG     = 4 * H;        // 800 real gate columns
constexpr int NGP    = 896;          // padded to 56 N-tiles -> exactly 7 tiles per wave
constexpr int XK     = 1152;         // padded K: 784 img + 160 r + 200 h + 8 zero
constexpr int KTILES = XK / 32;      // 36
constexpr int NTILES = NG / 16;      // 50 real tiles
constexpr int TPW    = NGP / 16 / 8; // 7 tiles per wave, uniform
constexpr int BM     = 16;           // batch rows per block
constexpr float GAMMA = 0.95f;
constexpr int PD1    = D + 1;        // 41
constexpr int PCOLS  = R * PD1;      // 164
constexpr int RH     = R * D + H;    // 360 recurrent K elements
constexpr int RHCHUNKS = (R * D + H) * 2 / 16;  // 45 16-byte chunks per row (320B r + 400B h)

// ---- dynamic LDS arena (byte offsets); regions reused across phases ----
constexpr int SM_A     = 0;              // bf16 [16][1152] (36864)  -> later: w_r f32[16*4*128] (32768)
constexpr int SM_WR    = 0;
constexpr int SM_GATES = 36864;          // f32 [16][800]   (51200)  -> later regions below
constexpr int SM_WW    = 36864;          // f32 [16*4*128]  (32768)
constexpr int SM_MINV  = 36864 + 32768;  // f32 [16*128]    (8192)
constexpr int SM_LU    = 36864 + 40960;  // int [16*4]      (256)
constexpr int SM_WU    = 36864 + 41216;  // f32 [16*128]    (8192)
constexpr int SM_H     = 88064;          // f32 [16][200]   (12800)
constexpr int SM_P     = 100864;         // f32 [16][164]   (10496)  -> later: r_new f32[16*160]
constexpr int SM_RNEW  = 100864;
constexpr int SM_K     = 111360;         // f32 [16*4*40]   (10240)
constexpr int SM_SIG   = 121600;         // f32 [64]
constexpr int SM_KINV  = 121856;         // f32 [64]
constexpr int SMEM_BYTES = 122112;

// ---- workspace layout (byte offsets) ----
constexpr size_t OFF_WT = 0;                                   // bf16 [896][1152]
constexpr size_t OFF_M  = OFF_WT + (size_t)NGP * XK * 2;       // f32  [B][N][D]
constexpr size_t OFF_RB = OFF_M  + (size_t)B * N * D * 4;      // bf16 [B][R*D]
constexpr size_t OFF_HB = OFF_RB + (size_t)B * R * D * 2;      // bf16 [B][H]
constexpr size_t OFF_C  = OFF_HB + (size_t)B * H * 2;          // f32  [B][H]
constexpr size_t OFF_WU = OFF_C  + (size_t)B * H * 4;          // f32  [B][N]
constexpr size_t OFF_WRS= OFF_WU + (size_t)B * N * 4;          // f32  [B][R][N]
constexpr size_t OFF_PT = OFF_WRS+ (size_t)B * R * N * 4;      // f32  [200][2] partials
} // namespace

__device__ __forceinline__ float sigmf(float x) { return 1.0f / (1.0f + expf(-x)); }

// A fragment (bf16 16x32): lane<16 -> M=lane, K=0..15; lane>=16 -> M=lane-16, K=16..31
__device__ __forceinline__ v16bf load_a_frag(const bf16_t* a_sh, int kt, int lane) {
    int m = lane & 15;
    int kb = (lane < 16) ? 0 : 16;
    const bf16_t* row = a_sh + m * XK + kt * 32;
    v16bf a;
#pragma unroll
    for (int e = 0; e < 16; ++e) a[e] = row[kb + e];
    return a;
}

// B fragment (bf16 32x16) from pre-transposed WT[n][k]: lane reads 16 contiguous bf16
__device__ __forceinline__ v16bf load_b_frag(const bf16_t* __restrict__ WT, int nt, int kt, int lane) {
    int n = lane & 15;
    int kb = (lane < 16) ? 0 : 16;
    const bf16_t* p = WT + (size_t)(nt * 16 + n) * XK + kt * 32 + kb;
    v16bf b;
#pragma unroll
    for (int e = 0; e < 16; ++e) b[e] = p[e];
    return b;
}

// Build fused transposed bf16 weight: K = [784 img | 160 r | 200 h | 8 pad]; cols >= 800 zero
__global__ void k_prep(const float* __restrict__ Wx, const float* __restrict__ Wh,
                       bf16_t* __restrict__ WT) {
    int i = blockIdx.x * blockDim.x + threadIdx.x;
    if (i >= NGP * XK) return;
    int n = i / XK, k = i % XK;
    float v = 0.0f;
    if (n < NG) {
        if (k < IN)                 v = Wx[(size_t)k * NG + n];
        else if (k < IN + R * D)    v = Wx[(size_t)(IN + C + (k - IN)) * NG + n];
        else if (k < IN + R * D + H)v = Wh[(size_t)(k - IN - R * D) * NG + n];
    }
    WT[(size_t)n * XK + k] = (bf16_t)v;
}

__global__ void k_init(float* M, bf16_t* r, bf16_t* h, float* c, float* wu, float* wr) {
    size_t i = (size_t)blockIdx.x * blockDim.x + threadIdx.x;
    if (i < (size_t)B * N * D) M[i] = 1e-6f;
    if (i < (size_t)B * R * D) r[i] = (bf16_t)1e-6f;
    if (i < (size_t)B * H)     { h[i] = (bf16_t)0.0f; c[i] = 0.0f; }
    if (i < (size_t)B * N)     wu[i] = ((i % N) == 0) ? 1.0f : 0.0f;
    if (i < (size_t)B * R * N) wr[i] = ((i % N) == 0) ? 1.0f : 0.0f;
}

__global__ __launch_bounds__(256) void k_step(
    const float* __restrict__ images, const int* __restrict__ labels,
    const float* __restrict__ Wx, const float* __restrict__ b_lstm,
    const float* __restrict__ Wp, const float* __restrict__ bp,
    const float* __restrict__ Wc, const float* __restrict__ bc,
    const bf16_t* __restrict__ WT,
    float* __restrict__ Mst, bf16_t* __restrict__ rbf, bf16_t* __restrict__ hbf,
    float* __restrict__ cst, float* __restrict__ wust, float* __restrict__ wrst,
    float* __restrict__ out, int t) {
    extern __shared__ char smem[];
    bf16_t* a_sh    = (bf16_t*)(smem + SM_A);
    float*  w_r_sh  = (float*)(smem + SM_WR);
    float*  gates_sh= (float*)(smem + SM_GATES);
    float*  w_w_sh  = (float*)(smem + SM_WW);
    float*  Minv_sh = (float*)(smem + SM_MINV);
    int*    lu_sh   = (int*)  (smem + SM_LU);
    float*  wu_sh   = (float*)(smem + SM_WU);
    float*  h_sh    = (float*)(smem + SM_H);
    float*  p_sh    = (float*)(smem + SM_P);
    float*  rnew_sh = (float*)(smem + SM_RNEW);
    float*  k_sh    = (float*)(smem + SM_K);
    float*  sig_sh  = (float*)(smem + SM_SIG);
    float*  kinv_sh = (float*)(smem + SM_KINV);
    __shared__ int   lbl_sh[BM];
    __shared__ float logit_s[BM * C];

    const int tid = threadIdx.x;
    const int b0  = blockIdx.x * BM;

    // ---- Phase 0: stage [x_t, r, h] as bf16 in LDS ----
    if (tid < BM) {
        lbl_sh[tid] = (t > 0) ? labels[(size_t)(b0 + tid) * T + (t - 1)] : 0;
        // zero the 8-element (16B) K-pad of this row (arena is clobbered every step)
        *(uint4*)(smem + SM_A + tid * (XK * 2) + (IN + RH) * 2) = make_uint4(0u, 0u, 0u, 0u);
    }
    // image segment: f32 -> bf16 convert while staging
    for (int idx = tid; idx < BM * IN; idx += 256) {
        int m = idx / IN, k = idx % IN;
        a_sh[m * XK + k] = (bf16_t)images[((size_t)(b0 + m) * T + t) * IN + k];
    }
#if USE_ASYNC_LDS
    // recurrent segment: pure bf16 byte-copy -> async global->LDS DMA (ASYNCcnt path)
    for (int cidx = tid; cidx < BM * RHCHUNKS; cidx += 256) {
        int m = cidx / RHCHUNKS, s = cidx % RHCHUNKS;
        int b = b0 + m;
        char* g;
        int loff;
        if (s < (R * D * 2) / 16) {             // 20 chunks of r (320 B)
            g    = (char*)(rbf + (size_t)b * (R * D)) + s * 16;
            loff = SM_A + m * (XK * 2) + IN * 2 + s * 16;
        } else {                                 // 25 chunks of h (400 B)
            int sh = s - (R * D * 2) / 16;
            g    = (char*)(hbf + (size_t)b * H) + sh * 16;
            loff = SM_A + m * (XK * 2) + (IN + R * D) * 2 + sh * 16;
        }
        __builtin_amdgcn_global_load_async_to_lds_b128(
            (gv4i_p)g, (lv4i_p)(smem + loff), 0, 0);
    }
    asm volatile("s_wait_asynccnt 0x0" ::: "memory");
#else
    for (int idx = tid; idx < BM * RH; idx += 256) {
        int m = idx / RH, k = idx % RH;
        bf16_t v = (k < R * D) ? rbf[(size_t)(b0 + m) * (R * D) + k]
                               : hbf[(size_t)(b0 + m) * H + (k - R * D)];
        a_sh[m * XK + IN + k] = v;
    }
#endif
    __syncthreads();

    // ---- Phase 1: gates[16x800] = a @ WT^T via WMMA bf16 (f32 accum) ----
    // NGP = 896 columns -> 56 tiles -> exactly 7 per wave: straight-line inner loop,
    // all 7 B-fragments issued as one load clause, WMMAs overlap next iteration's loads.
    {
        const int wave = __builtin_amdgcn_readfirstlane(tid) >> 5;  // SGPR-uniform wave id
        const int lane = tid & 31;
        const int mbase = (lane >> 4) ? 8 : 0;
        v8f acc[TPW];
#pragma unroll
        for (int i = 0; i < TPW; ++i)
#pragma unroll
            for (int e = 0; e < 8; ++e) acc[i][e] = 0.0f;
        for (int kt = 0; kt < KTILES; ++kt) {
            v16bf af = load_a_frag(a_sh, kt, lane);
            v16bf bfrag[TPW];
#pragma unroll
            for (int i = 0; i < TPW; ++i) bfrag[i] = load_b_frag(WT, wave + 8 * i, kt, lane);
#pragma unroll
            for (int i = 0; i < TPW; ++i)
                acc[i] = __builtin_amdgcn_wmma_f32_16x16x32_bf16(
                    false, af, false, bfrag[i], (short)0, acc[i], false, false);
        }
        int n = lane & 15;
#pragma unroll
        for (int i = 0; i < TPW; ++i) {
            int nt = wave + 8 * i;
            if (nt < NTILES) {                // scalar guard: padded tiles are discarded
                int col = nt * 16 + n;
#pragma unroll
                for (int v = 0; v < 8; ++v) {
                    int m = mbase + v;
                    float g = acc[i][v] + b_lstm[col];
                    if (t > 0) g += Wx[(size_t)(IN + lbl_sh[m]) * NG + col]; // one-hot offset label
                    gates_sh[m * NG + col] = g;
                }
            }
        }
    }
    __syncthreads();

    // ---- Phase 2: LSTM elementwise ----
    for (int idx = tid; idx < BM * H; idx += 256) {
        int m = idx / H, j = idx % H;
        int b = b0 + m;
        float gi = gates_sh[m * NG + j];
        float gf = gates_sh[m * NG + H + j];
        float gg = gates_sh[m * NG + 2 * H + j];
        float go = gates_sh[m * NG + 3 * H + j];
        float cn = sigmf(gf) * cst[(size_t)b * H + j] + sigmf(gi) * tanhf(gg);
        float hn = sigmf(go) * tanhf(cn);
        cst[(size_t)b * H + j] = cn;
        hbf[(size_t)b * H + j] = (bf16_t)hn;   // bf16 state: only ever feeds next-step bf16 GEMM
        h_sh[m * H + j] = hn;                  // f32 copy for this step's Wp / Wc products
    }
    __syncthreads();

    // ---- Phase 3: p = h_new @ Wp + bp  (16x164, K=200) ----
    for (int idx = tid; idx < BM * PCOLS; idx += 256) {
        int m = idx / PCOLS, col = idx % PCOLS;
        float s = bp[col];
        const float* h = h_sh + m * H;
        for (int k = 0; k < H; ++k) s += h[k] * Wp[(size_t)k * PCOLS + col];
        p_sh[idx] = s;
    }
    __syncthreads();

    // ---- Phase 3b: keys (tanh) + inv-norm + sigma; Phase 4: memory row inv-norms ----
    if (tid < BM * R) {
        int m = tid >> 2, r = tid & 3;
        const float* pp = p_sh + m * PCOLS + r * PD1;
        float ss = 0.0f;
        float* kd = k_sh + (m * R + r) * D;
        for (int d = 0; d < D; ++d) { float kv = tanhf(pp[d]); kd[d] = kv; ss += kv * kv; }
        kinv_sh[tid] = 1.0f / (sqrtf(ss) + 1e-8f);
        sig_sh[tid]  = sigmf(pp[D]);
    }
    for (int pr = tid; pr < BM * N; pr += 256) {
        int m = pr >> 7, n = pr & 127;
        const float* Mr = Mst + ((size_t)(b0 + m) * N + n) * D;
        float ss = 0.0f;
        for (int d = 0; d < D; ++d) ss += Mr[d] * Mr[d];
        Minv_sh[pr] = 1.0f / (sqrtf(ss) + 1e-8f);
    }
    // ---- Phase 5: least-used top-R (stable tie-break like top_k) ----
    if (tid < BM) {
        int m = tid, b = b0 + m;
        float* wrow = wu_sh + m * N;
        for (int n = 0; n < N; ++n) wrow[n] = wust[(size_t)b * N + n];
        int chosen[R];
        for (int j = 0; j < R; ++j) {
            float best = 3.4e38f; int bi = 0;
            for (int n = 0; n < N; ++n) {
                bool used = false;
                for (int q = 0; q < j; ++q) used |= (chosen[q] == n);
                if (!used && wrow[n] < best) { best = wrow[n]; bi = n; }
            }
            chosen[j] = bi; lu_sh[m * R + j] = bi;
        }
    }
    __syncthreads();

    // ---- Phase 6: cosine content addressing + softmax -> w_r; w_w; update wr state ----
    if (tid < BM * R) {
        int m = tid >> 2, r = tid & 3;
        int b = b0 + m;
        const float* kr = k_sh + (m * R + r) * D;
        float kiv = kinv_sh[m * R + r];
        float* wrow = w_r_sh + (m * R + r) * N;
        float mx = -3.4e38f;
        for (int n = 0; n < N; ++n) {
            const float* Mr = Mst + ((size_t)b * N + n) * D;
            float dot = 0.0f;
            for (int d = 0; d < D; ++d) dot += kr[d] * Mr[d];
            float s = kiv * Minv_sh[m * N + n] * dot;
            wrow[n] = s; mx = fmaxf(mx, s);
        }
        float se = 0.0f;
        for (int n = 0; n < N; ++n) { float e = expf(wrow[n] - mx); wrow[n] = e; se += e; }
        float inv = 1.0f / se;
        float sg = sig_sh[m * R + r];
        const int* lu = lu_sh + m * R;
        float* wwrow = w_w_sh + (m * R + r) * N;
        float* wrold = wrst + ((size_t)b * R + r) * N;
        for (int n = 0; n < N; ++n) {
            float wr = wrow[n] * inv; wrow[n] = wr;
            float wlu = (n == lu[0] || n == lu[1] || n == lu[2] || n == lu[3]) ? 1.0f : 0.0f;
            wwrow[n] = sg * wrold[n] + (1.0f - sg) * wlu;
        }
        for (int n = 0; n < N; ++n) wrold[n] = wrow[n];
    }
    __syncthreads();

    // ---- Phase 7: usage update; Phase 8: erase LRU slot + additive write ----
    for (int pr = tid; pr < BM * N; pr += 256) {
        int m = pr >> 7, n = pr & 127;
        float s = GAMMA * wu_sh[m * N + n];
        for (int r = 0; r < R; ++r) s += w_r_sh[(m * R + r) * N + n] + w_w_sh[(m * R + r) * N + n];
        wust[(size_t)(b0 + m) * N + n] = s;
    }
    for (int pr = tid; pr < BM * N; pr += 256) {
        int m = pr >> 7, n = pr & 127;
        float keep = (n == lu_sh[m * R + 3]) ? 0.0f : 1.0f;
        float ww0 = w_w_sh[(m * R + 0) * N + n], ww1 = w_w_sh[(m * R + 1) * N + n];
        float ww2 = w_w_sh[(m * R + 2) * N + n], ww3 = w_w_sh[(m * R + 3) * N + n];
        const float* k0 = k_sh + (m * R + 0) * D; const float* k1 = k_sh + (m * R + 1) * D;
        const float* k2 = k_sh + (m * R + 2) * D; const float* k3 = k_sh + (m * R + 3) * D;
        float* Mr = Mst + ((size_t)(b0 + m) * N + n) * D;
        for (int d = 0; d < D; ++d)
            Mr[d] = keep * Mr[d] + ww0 * k0[d] + ww1 * k1[d] + ww2 * k2[d] + ww3 * k3[d];
    }
    __syncthreads();

    // ---- Phase 9: read vectors from updated memory ----
    for (int idx = tid; idx < BM * R * D; idx += 256) {
        int m = idx / (R * D); int rm = idx % (R * D);
        int b = b0 + m;
        const float* wr = w_r_sh + (m * R + rm / D) * N;
        const float* Mb = Mst + (size_t)b * N * D + (rm % D);
        float s = 0.0f;
        for (int n = 0; n < N; ++n) s += wr[n] * Mb[(size_t)n * D];
        rnew_sh[m * (R * D) + rm] = s;
        rbf[(size_t)b * (R * D) + rm] = (bf16_t)s;   // bf16 state for next-step GEMM
    }
    __syncthreads();

    // ---- Phase 10: classifier + softmax -> probs ----
    if (tid < BM * C) {
        int m = tid / C, c = tid % C;
        float s = bc[c];
        const float* h = h_sh + m * H;
        for (int k = 0; k < H; ++k) s += h[k] * Wc[(size_t)k * C + c];
        const float* rn = rnew_sh + m * (R * D);
        for (int k = 0; k < R * D; ++k) s += rn[k] * Wc[(size_t)(H + k) * C + c];
        logit_s[m * C + c] = s;
    }
    __syncthreads();
    if (tid < BM) {
        int m = tid, b = b0 + m;
        const float* l = logit_s + m * C;
        float mx = l[0];
        for (int c = 1; c < C; ++c) mx = fmaxf(mx, l[c]);
        float e[C]; float se = 0.0f;
        for (int c = 0; c < C; ++c) { e[c] = expf(l[c] - mx); se += e[c]; }
        float inv = 1.0f / se;
        for (int c = 0; c < C; ++c) out[((size_t)b * T + t) * C + c] = e[c] * inv;
    }
}

__global__ void k_reduce1(const float* __restrict__ probs, const int* __restrict__ labels,
                          float* __restrict__ partial) {
    __shared__ float sl[256], sc[256];
    int row = blockIdx.x * 256 + threadIdx.x;
    const float* p = probs + (size_t)row * C;
    float mx = p[0];
    for (int c = 1; c < C; ++c) mx = fmaxf(mx, p[c]);
    float se = 0.0f;
    for (int c = 0; c < C; ++c) se += expf(p[c] - mx);
    float lse = mx + logf(se);
    int lbl = labels[row];
    float loss = -(p[lbl] - lse);
    int pm = 0; float bv = p[0];
    for (int c = 1; c < C; ++c) if (p[c] > bv) { bv = p[c]; pm = c; }
    sl[threadIdx.x] = loss;
    sc[threadIdx.x] = (pm == lbl) ? 1.0f : 0.0f;
    __syncthreads();
    for (int s = 128; s > 0; s >>= 1) {
        if (threadIdx.x < s) { sl[threadIdx.x] += sl[threadIdx.x + s]; sc[threadIdx.x] += sc[threadIdx.x + s]; }
        __syncthreads();
    }
    if (threadIdx.x == 0) { partial[blockIdx.x * 2] = sl[0]; partial[blockIdx.x * 2 + 1] = sc[0]; }
}

__global__ void k_reduce2(const float* __restrict__ partial, float* __restrict__ out_tail) {
    if (threadIdx.x == 0 && blockIdx.x == 0) {
        float ls = 0.0f, cs = 0.0f;
        for (int i = 0; i < (B * T) / 256; ++i) { ls += partial[2 * i]; cs += partial[2 * i + 1]; }
        out_tail[0] = ls / (float)(B * T);
        out_tail[1] = cs / (float)(B * T);
    }
}

extern "C" void kernel_launch(void* const* d_in, const int* in_sizes, int n_in,
                              void* d_out, int out_size, void* d_ws, size_t ws_size,
                              hipStream_t stream) {
    (void)in_sizes; (void)n_in; (void)out_size; (void)ws_size;
    const float* images = (const float*)d_in[0];
    const int*   labels = (const int*)d_in[1];
    const float* Wx     = (const float*)d_in[2];
    const float* Wh     = (const float*)d_in[3];
    const float* b_lstm = (const float*)d_in[4];
    const float* Wp     = (const float*)d_in[5];
    const float* bp     = (const float*)d_in[6];
    const float* Wc     = (const float*)d_in[7];
    const float* bc     = (const float*)d_in[8];
    float* out = (float*)d_out;
    char*  ws  = (char*)d_ws;

    bf16_t* WT  = (bf16_t*)(ws + OFF_WT);
    float*  Mst = (float*) (ws + OFF_M);
    bf16_t* rbf = (bf16_t*)(ws + OFF_RB);
    bf16_t* hbf = (bf16_t*)(ws + OFF_HB);
    float*  cst = (float*) (ws + OFF_C);
    float*  wust= (float*) (ws + OFF_WU);
    float*  wrst= (float*) (ws + OFF_WRS);
    float*  part= (float*) (ws + OFF_PT);

    k_prep<<<(NGP * XK + 255) / 256, 256, 0, stream>>>(Wx, Wh, WT);
    k_init<<<(B * N * D + 255) / 256, 256, 0, stream>>>(Mst, rbf, hbf, cst, wust, wrst);
    for (int t = 0; t < T; ++t) {
        k_step<<<B / BM, 256, SMEM_BYTES, stream>>>(images, labels, Wx, b_lstm, Wp, bp, Wc, bc,
                                                    WT, Mst, rbf, hbf, cst, wust, wrst, out, t);
    }
    k_reduce1<<<(B * T) / 256, 256, 0, stream>>>(out, labels, part);
    k_reduce2<<<1, 32, 0, stream>>>(part, out + (size_t)B * T * C);
}
